// Seq2Seq_64149631533601
// MI455X (gfx1250) — matrix operand order repaired and checked by reading, not compiled, not run on hardware
//
#include <hip/hip_runtime.h>

// ---------------------------------------------------------------------------
// Types
// ---------------------------------------------------------------------------
typedef unsigned short u16;
typedef __attribute__((ext_vector_type(8)))  u16    v8u;
typedef __attribute__((ext_vector_type(16))) u16    v16u;
typedef __attribute__((ext_vector_type(16))) __bf16 v16bf;
typedef __attribute__((ext_vector_type(8)))  float  v8f;

#define Bsz 64
#define Sln 512
#define Fdim 256
#define Hdim 1024
#define Gdim 4096   // 4*H
#define Tlen 256

#define KBLK 256          // K-block staged in LDS
#define LDA_S 264         // padded LDS row stride (u16 elems): kills bank conflicts

__device__ __forceinline__ u16 f32_to_bf16(float f) {
    unsigned int u = __float_as_uint(f);
    unsigned int r = (u + 0x7FFFu + ((u >> 16) & 1u)) >> 16;   // RNE
    return (u16)r;
}

__device__ __forceinline__ v16bf mk16(v8u lo, v8u hi) {
    v16u t;
#pragma unroll
    for (int i = 0; i < 8; ++i) { t[i] = lo[i]; t[8 + i] = hi[i]; }
    return __builtin_bit_cast(v16bf, t);
}

__device__ __forceinline__ float sigm(float x) {
    return 1.0f / (1.0f + __expf(-x));
}

// ---------------------------------------------------------------------------
// f32 -> bf16 conversion
// ---------------------------------------------------------------------------
__global__ void cvt_bf16(const float* __restrict__ in, u16* __restrict__ out, int n) {
    int i = blockIdx.x * blockDim.x + threadIdx.x;
    if (i < n) out[i] = f32_to_bf16(in[i]);
}

__global__ void add_bias(const float* __restrict__ a, const float* __restrict__ b,
                         float* __restrict__ out, int n) {
    int i = blockIdx.x * blockDim.x + threadIdx.x;
    if (i < n) out[i] = a[i] + b[i];
}

// ---------------------------------------------------------------------------
// Core WMMA panel: acc[64x16 strip] += A[64 x K] @ W^T[K x 16-strip]
//   A staged in LDS per 256-wide K block, shared by all 8 waves of the block;
//   weights stream from L2 (global_load_b128 + global_prefetch_b8).
//   K must be a multiple of KBLK (true here: 256 or 1024).
// ---------------------------------------------------------------------------
__device__ __forceinline__ void mma_panel(const u16* __restrict__ A, int lda, int K,
                                          const u16* __restrict__ W,
                                          u16* __restrict__ As, int tid,
                                          int r, int half, int ncol, v8f acc[4]) {
    const u16* wrow = W + (size_t)(ncol + r) * (size_t)K;
    for (int kb = 0; kb < K; kb += KBLK) {
        __syncthreads();   // protect LDS reads of previous block
        // cooperative stage: 64 rows x 256 cols bf16 (32KB) -> LDS, 8 x v8u per thread
#pragma unroll
        for (int it = 0; it < 8; ++it) {
            int chunk = it * 256 + tid;          // 0..2047, 8 elems each
            int row   = chunk >> 5;              // 0..63  (32 chunks per row)
            int coli  = (chunk & 31) * 8;        // 0..248
            *(v8u*)(As + row * LDA_S + coli) =
                *(const v8u*)(A + (size_t)row * (size_t)lda + kb + coli);
        }
        __syncthreads();
#pragma unroll 2
        for (int k0 = 0; k0 < KBLK; k0 += 32) {
            // B fragment (32x16): v0..3 = K half*8+0..7, v4..7 = 16+half*8+0..7
            const u16* wp = wrow + kb + k0;
            v8u blo = *(const v8u*)(wp + half * 8);
            v8u bhi = *(const v8u*)(wp + 16 + half * 8);
            __builtin_prefetch(wrow + kb + k0 + 64, 0, 0);   // global_prefetch_b8 (GL2)
            v16bf bf = mk16(blo, bhi);
#pragma unroll
            for (int m = 0; m < 4; ++m) {
                // A fragment (16x32) from LDS: v0..3 = K half*8+0..7, v4..7 = 16+half*8+0..7
                const u16* ap = As + (m * 16 + r) * LDA_S + k0;
                v8u alo = *(const v8u*)(ap + half * 8);
                v8u ahi = *(const v8u*)(ap + 16 + half * 8);
                acc[m] = __builtin_amdgcn_wmma_f32_16x16x32_bf16(
                    false, mk16(alo, ahi), false, bf, (short)0, acc[m], false, false);
            }
        }
    }
}

// ---------------------------------------------------------------------------
// Panel-parallel gates GEMM (split-K across gridDim.y):
//   y==0: out0 = A1[64 x K1] @ W1^T      y==1: out1 = A2[64 x K2] @ W2^T
// Partial sums (no bias); bias + reduction fused into lstm_cell.
// ---------------------------------------------------------------------------
__global__ void gemm_gates2(const u16* __restrict__ A1, int lda1, int K1, const u16* __restrict__ W1,
                            const u16* __restrict__ A2, int lda2, int K2, const u16* __restrict__ W2,
                            float* __restrict__ out0, float* __restrict__ out1) {
    __shared__ u16 As[64 * LDA_S];           // 33 KB of the 320 KB WGP LDS

    const int tid   = threadIdx.x;
    const int lane  = tid & 31;
    const int wave  = tid >> 5;
    const int strip = blockIdx.x * (blockDim.x >> 5) + wave;
    const int ncol  = strip * 16;
    const int r     = lane & 15;
    const int half  = lane >> 4;

    const u16* A; int lda, K; const u16* W; float* out;
    if (blockIdx.y == 0) { A = A1; lda = lda1; K = K1; W = W1; out = out0; }
    else                 { A = A2; lda = lda2; K = K2; W = W2; out = out1; }

    v8f acc[4];
#pragma unroll
    for (int m = 0; m < 4; ++m)
#pragma unroll
        for (int j = 0; j < 8; ++j) acc[m][j] = 0.0f;

    mma_panel(A, lda, K, W, As, tid, r, half, ncol, acc);

    // D layout: lane r = column, VGPR j = row (j + half*8) within each 16x16 tile
#pragma unroll
    for (int m = 0; m < 4; ++m)
#pragma unroll
        for (int j = 0; j < 8; ++j) {
            int row = m * 16 + half * 8 + j;
            out[(size_t)row * Gdim + ncol + r] = acc[m][j];
        }
}

// ---------------------------------------------------------------------------
// Single-panel GEMM with bias (projection): out[64 x N] = A @ W^T + bias
// Optional bf16 copy (y_prev writeback), row stride ldy.
// ---------------------------------------------------------------------------
__global__ void gemm_bias(const u16* __restrict__ A, int lda, int K, const u16* __restrict__ W,
                          const float* __restrict__ bias,
                          float* __restrict__ out, int ldo,
                          u16* __restrict__ yprev, int ldy) {
    __shared__ u16 As[64 * LDA_S];

    const int tid   = threadIdx.x;
    const int lane  = tid & 31;
    const int wave  = tid >> 5;
    const int strip = blockIdx.x * (blockDim.x >> 5) + wave;
    const int ncol  = strip * 16;
    const int r     = lane & 15;
    const int half  = lane >> 4;

    const float bval = bias[ncol + r];
    v8f acc[4];
#pragma unroll
    for (int m = 0; m < 4; ++m)
#pragma unroll
        for (int j = 0; j < 8; ++j) acc[m][j] = bval;

    mma_panel(A, lda, K, W, As, tid, r, half, ncol, acc);

#pragma unroll
    for (int m = 0; m < 4; ++m)
#pragma unroll
        for (int j = 0; j < 8; ++j) {
            int row = m * 16 + half * 8 + j;
            float v = acc[m][j];
            out[(size_t)row * (size_t)ldo + ncol + r] = v;
            if (yprev) yprev[row * ldy + ncol + r] = f32_to_bf16(v);
        }
}

// ---------------------------------------------------------------------------
// LSTM cell elementwise: gates = g0 + g1 + bias, PyTorch order (i,f,g,o).
// Updates c (f32) and writes h as bf16.
// ---------------------------------------------------------------------------
__global__ void lstm_cell(const float* __restrict__ g0, const float* __restrict__ g1,
                          const float* __restrict__ bias,
                          float* __restrict__ c, u16* __restrict__ h) {
    int idx = blockIdx.x * blockDim.x + threadIdx.x;     // 0 .. 65535
    int b = idx >> 10;
    int j = idx & 1023;
    size_t base = (size_t)b * Gdim;
    float gi = sigm(g0[base + j]           + g1[base + j]           + bias[j]);
    float gf = sigm(g0[base + j + Hdim]    + g1[base + j + Hdim]    + bias[j + Hdim]);
    float gg = tanhf(g0[base + j + 2*Hdim] + g1[base + j + 2*Hdim]  + bias[j + 2*Hdim]);
    float go = sigm(g0[base + j + 3*Hdim]  + g1[base + j + 3*Hdim]  + bias[j + 3*Hdim]);
    float cn = gf * c[idx] + gi * gg;
    c[idx] = cn;
    h[idx] = f32_to_bf16(go * tanhf(cn));
}

// ---------------------------------------------------------------------------
// Host-side orchestration
// ---------------------------------------------------------------------------
extern "C" void kernel_launch(void* const* d_in, const int* in_sizes, int n_in,
                              void* d_out, int out_size, void* d_ws, size_t ws_size,
                              hipStream_t stream) {
    (void)in_sizes; (void)n_in; (void)out_size; (void)ws_size;

    const float* src      = (const float*)d_in[0];
    const float* eWih0f   = (const float*)d_in[1];
    const float* eWhh0f   = (const float*)d_in[2];
    const float* ebih0    = (const float*)d_in[3];
    const float* ebhh0    = (const float*)d_in[4];
    const float* eWih1f   = (const float*)d_in[5];
    const float* eWhh1f   = (const float*)d_in[6];
    const float* ebih1    = (const float*)d_in[7];
    const float* ebhh1    = (const float*)d_in[8];
    const float* dWih0f   = (const float*)d_in[9];
    const float* dWhh0f   = (const float*)d_in[10];
    const float* dbih0    = (const float*)d_in[11];
    const float* dbhh0    = (const float*)d_in[12];
    const float* dWih1f   = (const float*)d_in[13];
    const float* dWhh1f   = (const float*)d_in[14];
    const float* dbih1    = (const float*)d_in[15];
    const float* dbhh1    = (const float*)d_in[16];
    const float* projWf   = (const float*)d_in[17];
    const float* projb    = (const float*)d_in[18];
    float*       outf     = (float*)d_out;

    // ---- workspace carving (256B aligned) ----
    char* ws = (char*)d_ws;
    size_t off = 0;
    auto alloc = [&](size_t bytes) -> char* {
        char* p = ws + off;
        off += (bytes + 255) & ~(size_t)255;
        return p;
    };

    u16* src_b  = (u16*)alloc((size_t)Bsz * Sln * Fdim * 2);
    u16* eWih0  = (u16*)alloc((size_t)Gdim * Fdim * 2);
    u16* eWhh0  = (u16*)alloc((size_t)Gdim * Hdim * 2);
    u16* eWih1  = (u16*)alloc((size_t)Gdim * Hdim * 2);
    u16* eWhh1  = (u16*)alloc((size_t)Gdim * Hdim * 2);
    u16* dWih0  = (u16*)alloc((size_t)Gdim * Fdim * 2);
    u16* dWhh0  = (u16*)alloc((size_t)Gdim * Hdim * 2);
    u16* dWih1  = (u16*)alloc((size_t)Gdim * Hdim * 2);
    u16* dWhh1  = (u16*)alloc((size_t)Gdim * Hdim * 2);
    u16* projWb = (u16*)alloc((size_t)Fdim * Hdim * 2);
    float* be0  = (float*)alloc(Gdim * 4);
    float* be1  = (float*)alloc(Gdim * 4);
    float* bd0  = (float*)alloc(Gdim * 4);
    float* bd1  = (float*)alloc(Gdim * 4);
    float* g0p  = (float*)alloc((size_t)Bsz * Gdim * 4);   // panel-0 partial gates
    float* g1p  = (float*)alloc((size_t)Bsz * Gdim * 4);   // panel-1 partial gates
    float* c0   = (float*)alloc((size_t)Bsz * Hdim * 4);
    float* c1   = (float*)alloc((size_t)Bsz * Hdim * 4);
    u16* h0b    = (u16*)alloc((size_t)Bsz * Hdim * 2);
    u16* h1b    = (u16*)alloc((size_t)Bsz * Hdim * 2);
    u16* yprevb = (u16*)alloc((size_t)Bsz * Fdim * 2);

    auto cvt = [&](const float* in, u16* out, size_t n) {
        cvt_bf16<<<(unsigned)((n + 255) / 256), 256, 0, stream>>>(in, out, (int)n);
    };

    // ---- per-call conversions (deterministic) ----
    cvt(src,    src_b,  (size_t)Bsz * Sln * Fdim);
    cvt(eWih0f, eWih0,  (size_t)Gdim * Fdim);
    cvt(eWhh0f, eWhh0,  (size_t)Gdim * Hdim);
    cvt(eWih1f, eWih1,  (size_t)Gdim * Hdim);
    cvt(eWhh1f, eWhh1,  (size_t)Gdim * Hdim);
    cvt(dWih0f, dWih0,  (size_t)Gdim * Fdim);
    cvt(dWhh0f, dWhh0,  (size_t)Gdim * Hdim);
    cvt(dWih1f, dWih1,  (size_t)Gdim * Hdim);
    cvt(dWhh1f, dWhh1,  (size_t)Gdim * Hdim);
    cvt(projWf, projWb, (size_t)Fdim * Hdim);

    add_bias<<<Gdim / 256, 256, 0, stream>>>(ebih0, ebhh0, be0, Gdim);
    add_bias<<<Gdim / 256, 256, 0, stream>>>(ebih1, ebhh1, be1, Gdim);
    add_bias<<<Gdim / 256, 256, 0, stream>>>(dbih0, dbhh0, bd0, Gdim);
    add_bias<<<Gdim / 256, 256, 0, stream>>>(dbih1, dbhh1, bd1, Gdim);

    // ---- state init (per call: state carries through the stream) ----
    hipMemsetAsync(c0,     0, (size_t)Bsz * Hdim * 4, stream);
    hipMemsetAsync(c1,     0, (size_t)Bsz * Hdim * 4, stream);
    hipMemsetAsync(h0b,    0, (size_t)Bsz * Hdim * 2, stream);
    hipMemsetAsync(h1b,    0, (size_t)Bsz * Hdim * 2, stream);
    hipMemsetAsync(yprevb, 0, (size_t)Bsz * Fdim * 2, stream);

    const dim3 gGemm(Gdim / (16 * 8), 2);  // 32 strips-blocks x 2 panels = 64 WGs
    const dim3 gCell((Bsz * Hdim) / 256);
    const dim3 blk(256);

    // ---- encoder ----
    for (int t = 0; t < Sln; ++t) {
        gemm_gates2<<<gGemm, blk, 0, stream>>>(src_b + (size_t)t * Fdim, Sln * Fdim, Fdim, eWih0,
                                               h0b, Hdim, Hdim, eWhh0, g0p, g1p);
        lstm_cell<<<gCell, blk, 0, stream>>>(g0p, g1p, be0, c0, h0b);
        gemm_gates2<<<gGemm, blk, 0, stream>>>(h0b, Hdim, Hdim, eWih1,
                                               h1b, Hdim, Hdim, eWhh1, g0p, g1p);
        lstm_cell<<<gCell, blk, 0, stream>>>(g0p, g1p, be1, c1, h1b);
    }

    // ---- decoder (autoregressive) ----
    for (int t = 0; t < Tlen; ++t) {
        gemm_gates2<<<gGemm, blk, 0, stream>>>(yprevb, Fdim, Fdim, dWih0,
                                               h0b, Hdim, Hdim, dWhh0, g0p, g1p);
        lstm_cell<<<gCell, blk, 0, stream>>>(g0p, g1p, bd0, c0, h0b);
        gemm_gates2<<<gGemm, blk, 0, stream>>>(h0b, Hdim, Hdim, dWih1,
                                               h1b, Hdim, Hdim, dWhh1, g0p, g1p);
        lstm_cell<<<gCell, blk, 0, stream>>>(g0p, g1p, bd1, c1, h1b);
        // projection: y = h1 @ projW^T + proj_b ; f32 output + bf16 y_prev writeback
        gemm_bias<<<dim3(Fdim / (16 * 8)), blk, 0, stream>>>(
            h1b, Hdim, Hdim, projWb,
            projb, outf + (size_t)t * Fdim, Tlen * Fdim, yprevb, Fdim);
    }
}
